// TweetEmbedder_9723805958351
// MI455X (gfx1250) — matrix operand order repaired and checked by reading, not compiled, not run on hardware
//
#include <hip/hip_runtime.h>
#include <hip/hip_bf16.h>

typedef __attribute__((ext_vector_type(16))) _Float16 v16h;
typedef __attribute__((ext_vector_type(8)))  float    v8f;

#define NNODES   50000
#define NEDGES   800000
#define IN_DIM   256
#define HD       128     // H * D
#define NHEAD    4
#define NEG_SLOPE 0.2f
#define MTILES   (NNODES / 16)   // 3125, exact

// ---- monotonic float<->uint mapping so segment-max can use atomicMax(u32) ----
__device__ __forceinline__ unsigned f2mono(float f) {
    unsigned u = __float_as_uint(f);
    return (u & 0x80000000u) ? ~u : (u | 0x80000000u);
}
__device__ __forceinline__ float mono2f(unsigned u) {
    return (u & 0x80000000u) ? __uint_as_float(u & 0x7fffffffu)
                             : __uint_as_float(~u);
}

// ------------------------------------------------------------------
// Kernel 0: initialize output features, per-(node,head) max and denom
// ------------------------------------------------------------------
__global__ void gat_init(float* __restrict__ out_feat,
                         unsigned* __restrict__ mmax,
                         float* __restrict__ denom) {
    size_t i = (size_t)blockIdx.x * blockDim.x + threadIdx.x;
    if (i < (size_t)NNODES * HD)    out_feat[i] = 0.0f;
    if (i < (size_t)NNODES * NHEAD) { mmax[i] = 0u; denom[i] = 0.0f; }
}

// ------------------------------------------------------------------
// Kernel 1: el = h @ W_src + b_src ; er = h @ W_dst + b_dst  (WMMA)
//
// Block = 256 threads = 8 waves; each wave owns one 16-row m-tile and
// computes the full 16x128 output strip (8 column tiles, 8 K-steps ->
// 64 v_wmma per wave). W (256x128) is staged once per block into LDS,
// converted to f16 and pre-swizzled into exact WMMA B-fragment order
// so each B operand is one aligned 32B LDS vector load.
//
// Fragment layouts per CDNA5 ISA 7.12.2 (wave32):
//   A 16x32 f16 : lanes 0-15 row M=lane; halves = K {kh*8+0..7, kh*8+16..23}
//   B 32x16 f16 : lane holds col N=lane&15; halves = K {kh*8+0..7, kh*8+16..23}
//   C/D 16x16 f32: VGPR r -> row r + (lane>>4)*8, col = lane&15
// LDS fragment index (in v16h units): (kk*8 + nt)*32 + lane
// ------------------------------------------------------------------
__global__ __launch_bounds__(256) void gat_gemm(
    const float* __restrict__ h,
    const float* __restrict__ W_src, const float* __restrict__ b_src,
    const float* __restrict__ W_dst, const float* __restrict__ b_dst,
    float* __restrict__ el, float* __restrict__ er)
{
    __shared__ __align__(32) unsigned int smem[IN_DIM * HD / 2]; // 64 KB, f16 pairs

    const int which = blockIdx.y;           // 0 = src, 1 = dst
    const float* __restrict__ W    = which ? W_dst : W_src;
    const float* __restrict__ bias = which ? b_dst : b_src;
    float*       __restrict__ outb = which ? er    : el;

    // ---- stage W into LDS in B-fragment order (each u32 = 2 packed f16) ----
    // u32 index idx: j2 = idx&7 (half-pair), lane = (idx>>3)&31,
    //                nt = (idx>>8)&7, kk = idx>>11
#pragma unroll
    for (int i = 0; i < (IN_DIM * HD / 2) / 256; ++i) {
        const int idx  = threadIdx.x + 256 * i;
        const int j2   = idx & 7;
        const int lane = (idx >> 3) & 31;
        const int nt   = (idx >> 8) & 7;
        const int kk   = idx >> 11;
        const int k    = kk * 32 + (lane >> 4) * 8 + (j2 & 3) * 2 + (j2 >> 2) * 16;
        const int col  = nt * 16 + (lane & 15);
        union { _Float16 hh[2]; unsigned u; } pk;
        pk.hh[0] = (_Float16)W[(size_t)k       * HD + col];
        pk.hh[1] = (_Float16)W[(size_t)(k + 1) * HD + col];
        smem[idx] = pk.u;
    }
    __syncthreads();

    const int wave  = threadIdx.x >> 5;
    const int lane  = threadIdx.x & 31;
    const int mtile = blockIdx.x * 8 + wave;     // one m-tile per wave
    if (mtile < MTILES) {                         // wave-uniform branch
        const int l15 = lane & 15;
        const int kh  = lane >> 4;
        const float* hrow = h + (size_t)(mtile * 16 + l15) * IN_DIM;
        const v16h* frag = (const v16h*)smem;

        v8f c[8];
#pragma unroll
        for (int nt = 0; nt < 8; ++nt) c[nt] = (v8f){};

#pragma unroll
        for (int kk = 0; kk < IN_DIM / 32; ++kk) {
            v16h a;
            const int ka = kk * 32 + kh * 8;
#pragma unroll
            for (int j = 0; j < 8; ++j) {
                a[j]     = (_Float16)hrow[ka + j];
                a[8 + j] = (_Float16)hrow[ka + 16 + j];
            }
#pragma unroll
            for (int nt = 0; nt < 8; ++nt) {
                const v16h b = frag[(kk * 8 + nt) * 32 + lane];
                c[nt] = __builtin_amdgcn_wmma_f32_16x16x32_f16(
                            false, a, false, b, (short)0, c[nt], false, false);
            }
        }

#pragma unroll
        for (int nt = 0; nt < 8; ++nt) {
            const int col = nt * 16 + l15;
            const float bv = bias[col];
#pragma unroll
            for (int r = 0; r < 8; ++r) {
                const int row = mtile * 16 + kh * 8 + r;
                outb[(size_t)row * HD + col] = c[nt][r] + bv;
            }
        }
    }
}

// ------------------------------------------------------------------
// Kernel 2: per-edge scores + running segment max over dst.
// One wave per edge: lane holds 4 of the 128 features; head = lane>>3.
// ------------------------------------------------------------------
__global__ __launch_bounds__(256) void gat_scores_max(
    const float* __restrict__ el, const float* __restrict__ er,
    const int* __restrict__ src, const int* __restrict__ dst,
    const float* __restrict__ attn,
    float* __restrict__ scores, unsigned* __restrict__ mmax)
{
    const int e = blockIdx.x * 8 + (threadIdx.x >> 5);
    if (e >= NEDGES) return;
    const int lane = threadIdx.x & 31;
    const int s = src[e], d = dst[e];

    __builtin_prefetch(el + (size_t)s * HD, 0, 1);   // global_prefetch_b8
    const float4 a4 = ((const float4*)(el + (size_t)s * HD))[lane];
    const float4 b4 = ((const float4*)(er + (size_t)d * HD))[lane];
    const float4 w4 = ((const float4*)attn)[lane];

    float p = 0.0f, x;
    x = a4.x + b4.x; p += (x > 0.0f ? x : NEG_SLOPE * x) * w4.x;
    x = a4.y + b4.y; p += (x > 0.0f ? x : NEG_SLOPE * x) * w4.y;
    x = a4.z + b4.z; p += (x > 0.0f ? x : NEG_SLOPE * x) * w4.z;
    x = a4.w + b4.w; p += (x > 0.0f ? x : NEG_SLOPE * x) * w4.w;

    // reduce over the 8 lanes of each head
    p += __shfl_xor(p, 1, 32);
    p += __shfl_xor(p, 2, 32);
    p += __shfl_xor(p, 4, 32);

    if ((lane & 7) == 0) {
        const int hh = lane >> 3;
        scores[(size_t)e * NHEAD + hh] = p;
        atomicMax(mmax + (size_t)d * NHEAD + hh, f2mono(p));
    }
}

// ------------------------------------------------------------------
// Kernel 3: ex = exp(score - max[dst]); denom[dst] += ex  (in-place)
// ------------------------------------------------------------------
__global__ void gat_exp_denom(const int* __restrict__ dst,
                              float* __restrict__ scores,
                              const unsigned* __restrict__ mmax,
                              float* __restrict__ denom)
{
    const int i = blockIdx.x * blockDim.x + threadIdx.x;
    if (i >= NEDGES * NHEAD) return;
    const int e = i >> 2, hh = i & 3;
    const int d = dst[e];
    const float ex = expf(scores[i] - mono2f(mmax[(size_t)d * NHEAD + hh]));
    scores[i] = ex;                          // reuse buffer for ex
    atomicAdd(denom + (size_t)d * NHEAD + hh, ex);
}

// ------------------------------------------------------------------
// Kernel 4: a = ex/denom[dst]; out[dst] += a * el[src]; emit a.
// One wave per edge; 4 float atomics per lane -> global_atomic_add_f32.
// ------------------------------------------------------------------
__global__ __launch_bounds__(256) void gat_aggregate(
    const float* __restrict__ el,
    const int* __restrict__ src, const int* __restrict__ dst,
    const float* __restrict__ ex, const float* __restrict__ denom,
    float* __restrict__ out_feat, float* __restrict__ out_attn)
{
    const int e = blockIdx.x * 8 + (threadIdx.x >> 5);
    if (e >= NEDGES) return;
    const int lane = threadIdx.x & 31;
    const int s = src[e], d = dst[e];
    const int hh = lane >> 3;

    const float a = ex[(size_t)e * NHEAD + hh] /
                    denom[(size_t)d * NHEAD + hh];

    const float4 v = ((const float4*)(el + (size_t)s * HD))[lane];
    float* o = out_feat + (size_t)d * HD + lane * 4;
    atomicAdd(o + 0, a * v.x);
    atomicAdd(o + 1, a * v.y);
    atomicAdd(o + 2, a * v.z);
    atomicAdd(o + 3, a * v.w);

    if ((lane & 7) == 0) out_attn[(size_t)e * NHEAD + hh] = a;
}

// ------------------------------------------------------------------
extern "C" void kernel_launch(void* const* d_in, const int* in_sizes, int n_in,
                              void* d_out, int out_size, void* d_ws, size_t ws_size,
                              hipStream_t stream)
{
    const float* h     = (const float*)d_in[0];
    const int*   src   = (const int*)  d_in[1];
    const int*   dst   = (const int*)  d_in[2];
    const float* W_src = (const float*)d_in[3];
    const float* b_src = (const float*)d_in[4];
    const float* W_dst = (const float*)d_in[5];
    const float* b_dst = (const float*)d_in[6];
    const float* attn  = (const float*)d_in[7];

    float* out_feat = (float*)d_out;                       // (N, 128)
    float* out_attn = out_feat + (size_t)NNODES * HD;      // (E, 4)

    // workspace layout (floats): el | er | scores/ex | mmax(u32) | denom
    float*    el     = (float*)d_ws;
    float*    er     = el + (size_t)NNODES * HD;
    float*    scores = er + (size_t)NNODES * HD;
    unsigned* mmax   = (unsigned*)(scores + (size_t)NEDGES * NHEAD);
    float*    denom  = (float*)(mmax + (size_t)NNODES * NHEAD);

    gat_init<<<((size_t)NNODES * HD + 255) / 256, 256, 0, stream>>>(
        out_feat, mmax, denom);

    gat_gemm<<<dim3((MTILES + 7) / 8, 2), 256, 0, stream>>>(
        h, W_src, b_src, W_dst, b_dst, el, er);

    gat_scores_max<<<(NEDGES + 7) / 8, 256, 0, stream>>>(
        el, er, src, dst, attn, scores, mmax);

    gat_exp_denom<<<(NEDGES * NHEAD + 255) / 256, 256, 0, stream>>>(
        dst, scores, mmax, denom);

    gat_aggregate<<<(NEDGES + 7) / 8, 256, 0, stream>>>(
        el, src, dst, scores, denom, out_feat, out_attn);
}
